// Seq2Seq_29111288332698
// MI455X (gfx1250) — matrix operand order repaired and checked by reading, not compiled, not run on hardware
//
#include <hip/hip_runtime.h>
#include <hip/hip_bf16.h>

// ---------------------------------------------------------------------------
// Problem constants (match the reference)
// ---------------------------------------------------------------------------
#define BATCH 32
#define TSEQ  128
#define EDIM  512
#define UDIM  512
#define VDIM  9000

typedef __attribute__((ext_vector_type(16))) __bf16 v16bf;
typedef __attribute__((ext_vector_type(8)))  float  v8f;

union Frag16 { v16bf v; unsigned int w[8]; };

__device__ inline unsigned short f2bf(float x) {
  unsigned int u = __float_as_uint(x);
  u += 0x7FFFu + ((u >> 16) & 1u);              // round-to-nearest-even
  return (unsigned short)(u >> 16);
}
__device__ inline float sigm(float x) { return 1.0f / (1.0f + expf(-x)); }

__device__ inline v8f zero8() { v8f z = {0.f,0.f,0.f,0.f,0.f,0.f,0.f,0.f}; return z; }

__device__ inline v8f wmma_bf16(v16bf a, v16bf b, v8f c) {
  return __builtin_amdgcn_wmma_f32_16x16x32_bf16(false, a, false, b, (short)0, c, false, false);
}

// ---------------------------------------------------------------------------
// Tensor Data Mover (gfx1250): async 2D tile load Global->LDS with HW padding.
// D# layout per CDNA5 ISA.  Issued by one wave per
// block; completion tracked on TENSORcnt (in-order per wave).
// ---------------------------------------------------------------------------
#if __has_builtin(__builtin_amdgcn_tensor_load_to_lds) && \
    __has_builtin(__builtin_amdgcn_s_wait_tensorcnt)
#define USE_TDM 1
#else
#define USE_TDM 0
#endif

#if USE_TDM
typedef unsigned int u32x4 __attribute__((ext_vector_type(4)));
typedef int i32x4 __attribute__((ext_vector_type(4)));
typedef int i32x8 __attribute__((ext_vector_type(8)));

// Loads a [tile_d1][tile_d0] tile of 2-byte elements starting at gptr
// (row stride stride_d0 elements) into LDS at lds_addr, inserting
// pad_amount(+1) DWORDs of padding after each 2^(pad_interval+1) DWORDs.
__device__ inline void tdm_load_tile_2d(unsigned lds_addr, const void* gptr,
                                        int tensor_d0, int tensor_d1,
                                        int tile_d0, int tile_d1,
                                        int stride_d0,
                                        int pad_interval_code, int pad_amount_code) {
  unsigned long long ga = (unsigned long long)gptr;
  u32x4 g0;
  g0[0] = 1u;                                               // count=1: valid user D#
  g0[1] = lds_addr;                                         // LDS byte address
  g0[2] = (unsigned)ga;                                     // global_addr[31:0]
  g0[3] = (unsigned)((ga >> 32) & 0x1FFFFFFull) | (2u << 30); // addr[56:32] | type=2
  i32x8 g1;
  g1[0] = (1 << 16)                                         // data_size: 2 bytes
        | (1 << 20)                                         // pad_enable
        | (pad_interval_code << 22)
        | (pad_amount_code << 25);
  g1[1] = (tensor_d0 & 0xFFFF) << 16;                       // tensor_dim0[15:0]
  g1[2] = (int)(((unsigned)tensor_d0 >> 16) | (((unsigned)tensor_d1 & 0xFFFF) << 16));
  g1[3] = (int)((((unsigned)tensor_d1 >> 16) & 0xFFFF) | (((unsigned)tile_d0 & 0xFFFF) << 16));
  g1[4] = tile_d1 & 0xFFFF;                                 // tile_dim1 (tile_dim2=0)
  g1[5] = stride_d0;                                        // tensor_dim0_stride[31:0]
  g1[6] = 0;                                                // stride hi | dim1_stride lo
  g1[7] = 0;
  i32x4 z4 = {0, 0, 0, 0};
#if __clang_major__ >= 23
  i32x8 z8 = {0, 0, 0, 0, 0, 0, 0, 0};
  __builtin_amdgcn_tensor_load_to_lds(g0, g1, z4, z4, z8, 0);
#else
  __builtin_amdgcn_tensor_load_to_lds(g0, g1, z4, z4, 0);
#endif
}
#endif  // USE_TDM

// ---------------------------------------------------------------------------
// WMMA fragment loaders (CDNA5 16-bit VGPR layouts, CDNA5 ISA)
// A 16x32 bf16: lanes 0-15 -> M=lane, K {8h..8h+7, 16+8h..}, pairs per VGPR.
// B 32x16 bf16 (stored K-major: base[n*strideK + k]).
// C/D 16x16 f32: lane n = lane&15, VGPR j -> M = j + 8*(lane>>4).
// All strides are even elements => u32 LDS/global reads are DWORD-aligned.
// ---------------------------------------------------------------------------
__device__ inline v16bf load_a_frag(const unsigned short* base, int stride,
                                    int m0, int k0, int lane) {
  int m  = m0 + (lane & 15);
  int h8 = (lane >> 4) * 8;
  const unsigned short* row = base + m * stride + k0;
  Frag16 f;
#pragma unroll
  for (int j = 0; j < 4; j++) {
    f.w[j]     = *(const unsigned int*)(row + h8 + 2 * j);
    f.w[j + 4] = *(const unsigned int*)(row + 16 + h8 + 2 * j);
  }
  return f.v;
}

__device__ inline v16bf load_b_frag(const unsigned short* base, int strideK,
                                    int n0, int k0, int lane) {
  int n  = n0 + (lane & 15);
  int kh = (lane >> 4) * 8;
  const unsigned short* col = base + n * strideK + k0 + kh;
  Frag16 f;
#pragma unroll
  for (int j = 0; j < 8; j++)
    f.w[j] = *(const unsigned int*)(col + ((j < 4) ? (2 * j) : (16 + 2 * (j - 4))));
  return f.v;
}

// A fragment over the concatenation [bufA (k<split) | bufB (k>=split)]
__device__ inline v16bf load_a_frag_split(const unsigned short* bufA, int strideA,
                                          const unsigned short* bufB, int strideB,
                                          int split, int m0, int k0, int lane) {
  int m  = m0 + (lane & 15);
  int h8 = (lane >> 4) * 8;
  Frag16 f;
#pragma unroll
  for (int j = 0; j < 8; j++) {
    int k = k0 + ((j < 4) ? (h8 + 2 * j) : (16 + h8 + 2 * (j - 4)));
    const unsigned short* p = (k < split) ? (bufA + m * strideA + k)
                                          : (bufB + m * strideB + (k - split));
    f.w[j] = *(const unsigned int*)p;
  }
  return f.v;
}

// ---------------------------------------------------------------------------
// Weight conversion: fp32 -> bf16 K-major: dst[n*K + k] = src[(row0+k)*srcN + n]
// ---------------------------------------------------------------------------
__global__ void __launch_bounds__(256) f32_to_bf16_T_kernel(const float* __restrict__ src,
                                                            int srcN, int row0, int K, int N,
                                                            unsigned short* __restrict__ dst) {
  int total = K * N;
  for (int i = blockIdx.x * blockDim.x + threadIdx.x; i < total; i += gridDim.x * blockDim.x) {
    int n = i / K;
    int k = i - n * K;
    dst[i] = f2bf(src[(size_t)(row0 + k) * srcN + n]);
  }
}

// vn = g * v / sqrt(sum(v*v))
__global__ void __launch_bounds__(512) vnorm_kernel(const float* __restrict__ v,
                                                    const float* __restrict__ g,
                                                    float* __restrict__ vn) {
  __shared__ float red[512];
  int i = threadIdx.x;
  float x = v[i];
  red[i] = x * x;
  __syncthreads();
  for (int s = 256; s > 0; s >>= 1) {
    if (i < s) red[i] += red[i + s];
    __syncthreads();
  }
  vn[i] = g[0] * x * rsqrtf(red[0]);
}

// ---------------------------------------------------------------------------
// Generic bf16 WMMA GEMM:  C[M,N] = A[M,K](f32) @ B (bf16, K-major [N][K]) + bias
// Block: 256 threads = 8 waves, tile 64(M) x 128(N), K-step 32.
// B tiles are DMA'd into LDS by the Tensor Data Mover (double-buffered,
// HW-padded to a 36-element row stride); A is converted f32->bf16 in VALU.
// lens != null => A rows gathered with tf.reverse_sequence semantics.
// ---------------------------------------------------------------------------
__global__ void __launch_bounds__(256) gemm_bf16_kernel(
    const float* __restrict__ A, int K, int N,
    const unsigned short* __restrict__ B,     // K-major [N][K]
    const float* __restrict__ bias,
    float* __restrict__ C,
    const int* __restrict__ lens) {
  __shared__ unsigned short As[64 * 40];        // [m][k], stride 40 (80B rows)
  __shared__ unsigned short Bs[2][128 * 36];    // [n][k], stride 36 (72B rows)

  const int tid  = threadIdx.x;
  const int lane = tid & 31;
  const int wv   = tid >> 5;      // 0..7
  const int wm   = wv & 1;        // 2 waves across M
  const int wn   = wv >> 1;       // 4 waves across N
  const int mBase = blockIdx.y * 64;
  const int nBase = blockIdx.x * 128;

  v8f acc[2][2];
  acc[0][0] = zero8(); acc[0][1] = zero8();
  acc[1][0] = zero8(); acc[1][1] = zero8();

  // A-row gather (reverse_seq for the backward encoder)
  const int ar = tid >> 2;          // 0..63
  const int ac = (tid & 3) * 8;     // 0,8,16,24
  int srow;
  {
    int m = mBase + ar;
    if (lens) {
      int b = m >> 7, t = m & 127;
      int L = lens[b];
      int st = (t < L) ? (L - 1 - t) : t;
      srow = (b << 7) + st;
    } else {
      srow = m;
    }
  }

#if USE_TDM
  if (wv == 0) {   // prime the pipeline: tile kk=0 into buffer 0
    tdm_load_tile_2d((unsigned)(size_t)&Bs[0][0],
                     B + (size_t)nBase * K,
                     /*tensor_d0=*/K, /*tensor_d1=*/N - nBase,
                     /*tile_d0=*/32, /*tile_d1=*/128, /*stride_d0=*/K,
                     /*pad_interval: 16 DWORDs*/3, /*pad: 2 DWORDs*/1);
  }
#endif

  for (int kk = 0; kk < K; kk += 32) {
    const int buf = (kk >> 5) & 1;
#if USE_TDM
    if (kk + 32 < K) {             // issue next tile while we convert A
      if (wv == 0)
        tdm_load_tile_2d((unsigned)(size_t)&Bs[buf ^ 1][0],
                         B + (size_t)nBase * K + (kk + 32),
                         K - (kk + 32), N - nBase, 32, 128, K, 3, 1);
    }
#else
    // manual B staging fallback (zero-pad past N)
#pragma unroll
    for (int i = 0; i < 16; i++) {
      int lin = i * 256 + tid;
      int n = lin >> 5, k = lin & 31;
      int gn = nBase + n;
      unsigned short val = 0;
      if (gn < N) val = B[(size_t)gn * K + kk + k];
      Bs[buf][n * 36 + k] = val;
    }
#endif
    { // stage A 64x32 (f32 -> bf16)
      const float* ap = A + (size_t)srow * K + kk + ac;
      __builtin_prefetch(ap + 32, 0, 1);
      unsigned short* dp = &As[ar * 40 + ac];
#pragma unroll
      for (int i = 0; i < 8; i++) dp[i] = f2bf(ap[i]);
    }
#if USE_TDM
    if (kk + 32 < K) __builtin_amdgcn_s_wait_tensorcnt(1);  // tile kk complete
    else             __builtin_amdgcn_s_wait_tensorcnt(0);
#endif
    __syncthreads();

    v16bf a0 = load_a_frag(As, 40, wm * 32,      0, lane);
    v16bf a1 = load_a_frag(As, 40, wm * 32 + 16, 0, lane);
    v16bf b0 = load_b_frag(Bs[buf], 36, wn * 32,      0, lane);
    v16bf b1 = load_b_frag(Bs[buf], 36, wn * 32 + 16, 0, lane);
    acc[0][0] = wmma_bf16(a0, b0, acc[0][0]);
    acc[0][1] = wmma_bf16(a0, b1, acc[0][1]);
    acc[1][0] = wmma_bf16(a1, b0, acc[1][0]);
    acc[1][1] = wmma_bf16(a1, b1, acc[1][1]);
    __syncthreads();
  }

  const int nl = lane & 15, mh = lane >> 4;
#pragma unroll
  for (int mt = 0; mt < 2; mt++) {
#pragma unroll
    for (int nt = 0; nt < 2; nt++) {
      int n = nBase + wn * 32 + nt * 16 + nl;
      if (n < N) {
        float bv = bias ? bias[n] : 0.f;
        int m0 = mBase + wm * 32 + mt * 16 + mh * 8;
#pragma unroll
        for (int j = 0; j < 8; j++)
          C[(size_t)(m0 + j) * N + n] = acc[mt][nt][j] + bv;
      }
    }
  }
}

// ---------------------------------------------------------------------------
// Encoder recurrence: one workgroup (32 waves) per direction. Per step:
//   z = zx[t] + h @ Wh  (WMMA, K=512; wave w owns u-column [16w,16w+16) for
//   all four gates -> LSTM cell is pure register math; c lives in registers).
// Writes enc_out[B,T,2U] (backward half time-reversed in place).
// ---------------------------------------------------------------------------
__global__ void __launch_bounds__(1024) encoder_kernel(
    const float* __restrict__ zx_fw, const float* __restrict__ zx_bw,
    const unsigned short* __restrict__ Whr_fw,   // [2048][512] K-major bf16
    const unsigned short* __restrict__ Whr_bw,
    const int* __restrict__ lens,
    float* __restrict__ enc_out,
    float* __restrict__ c_fw, float* __restrict__ h_fw,
    float* __restrict__ c_bw, float* __restrict__ h_bw) {
  const int dir = blockIdx.x;                 // 0 = fw, 1 = bw
  const float* zx = dir ? zx_bw : zx_fw;
  const unsigned short* Whr = dir ? Whr_bw : Whr_fw;
  float* c_fin = dir ? c_bw : c_fw;
  float* h_fin = dir ? h_bw : h_fw;
  const int colofs = dir ? UDIM : 0;

  __shared__ unsigned short hbuf[BATCH * UDIM];   // bf16 h state, 32 KB
  __shared__ int lens_s[BATCH];

  const int tid  = threadIdx.x;
  const int lane = tid & 31;
  const int w    = tid >> 5;     // 32 waves
  const int u0   = w * 16;
  const int nl   = lane & 15;
  const int mh   = lane >> 4;

  for (int i = tid; i < BATCH * UDIM; i += 1024) hbuf[i] = 0;
  if (tid < BATCH) lens_s[tid] = lens[tid];

  float c_r[2][8], h_keep[2][8];
#pragma unroll
  for (int mt = 0; mt < 2; mt++)
#pragma unroll
    for (int j = 0; j < 8; j++) { c_r[mt][j] = 0.f; h_keep[mt][j] = 0.f; }

  for (int t = 0; t < TSEQ; t++) {
    __syncthreads();                              // h state visible
    v8f acc[4][2];
#pragma unroll
    for (int g = 0; g < 4; g++) { acc[g][0] = zero8(); acc[g][1] = zero8(); }

    for (int kk = 0; kk < UDIM; kk += 32) {
      v16bf a0 = load_a_frag(hbuf, UDIM, 0,  kk, lane);
      v16bf a1 = load_a_frag(hbuf, UDIM, 16, kk, lane);
#pragma unroll
      for (int g = 0; g < 4; g++) {
        v16bf bf = load_b_frag(Whr, UDIM, g * UDIM + u0, kk, lane);
        acc[g][0] = wmma_bf16(a0, bf, acc[g][0]);
        acc[g][1] = wmma_bf16(a1, bf, acc[g][1]);
      }
    }

#pragma unroll
    for (int mt = 0; mt < 2; mt++) {
#pragma unroll
      for (int j = 0; j < 8; j++) {
        int b = mt * 16 + mh * 8 + j;
        int zidx = ((b << 7) + t) * 2048 + u0 + nl;
        float gi = acc[0][mt][j] + zx[zidx];
        float gj = acc[1][mt][j] + zx[zidx + 512];
        float gf = acc[2][mt][j] + zx[zidx + 1024];
        float go = acc[3][mt][j] + zx[zidx + 1536];
        float cn = sigm(gf + 1.0f) * c_r[mt][j] + sigm(gi) * tanhf(gj);
        float hn = sigm(go) * tanhf(cn);
        int L = lens_s[b];
        bool valid = t < L;
        c_r[mt][j]    = valid ? cn : c_r[mt][j];
        h_keep[mt][j] = valid ? hn : h_keep[mt][j];
        int dst_t = dir ? ((t < L) ? (L - 1 - t) : t) : t;
        enc_out[((b << 7) + dst_t) * 1024 + colofs + u0 + nl] = valid ? hn : 0.f;
      }
    }
    __syncthreads();                              // all GEMM reads of hbuf done
#pragma unroll
    for (int mt = 0; mt < 2; mt++)
#pragma unroll
      for (int j = 0; j < 8; j++) {
        int b = mt * 16 + mh * 8 + j;
        hbuf[b * UDIM + u0 + nl] = f2bf(h_keep[mt][j]);
      }
  }

#pragma unroll
  for (int mt = 0; mt < 2; mt++)
#pragma unroll
    for (int j = 0; j < 8; j++) {
      int b = mt * 16 + mh * 8 + j;
      c_fin[b * UDIM + u0 + nl] = c_r[mt][j];
      h_fin[b * UDIM + u0 + nl] = h_keep[mt][j];
    }
}

// ---------------------------------------------------------------------------
// Decoder: one persistent workgroup, all state in LDS/registers.
// Per step: layer0 WMMA -> cell -> layer1 WMMA -> cell -> query WMMA ->
// score(tanh)+masked softmax+context (VALU, one wave per batch row) ->
// attention-layer WMMA -> new attn vector.
// ---------------------------------------------------------------------------
__global__ void __launch_bounds__(1024) decoder_kernel(
    const float* __restrict__ zx0,              // [B*T][2048] (x@W + b folded)
    const unsigned short* __restrict__ W0r,     // [2048][1024] K-major (attn|h0 rows)
    const unsigned short* __restrict__ W1r,     // [2048][1024] K-major (h0n|h1 rows)
    const float* __restrict__ bias1,            // [2048]
    const unsigned short* __restrict__ Wqr,     // [512][512]  K-major
    const unsigned short* __restrict__ Walr,    // [512][1536] K-major
    const float* __restrict__ keys,             // [B*T][512] (b_attn folded)
    const float* __restrict__ enc_out,          // [B*T][1024]
    const float* __restrict__ vn,               // [512]
    const int* __restrict__ lens,
    const float* __restrict__ c0i, const float* __restrict__ h0i,
    const float* __restrict__ c1i, const float* __restrict__ h1i,
    float* __restrict__ dec_out) {              // [B*T][512]
  extern __shared__ char smem[];
  unsigned short* attnB = (unsigned short*)smem;             // [32][512] bf16
  unsigned short* h0B   = attnB + BATCH * UDIM;              // [32][512]
  unsigned short* h1B   = h0B + BATCH * UDIM;                // [32][512]
  unsigned short* ctxB  = h1B + BATCH * UDIM;                // [32][1024]
  float* qS   = (float*)(ctxB + BATCH * 1024);               // [32][512]
  float* scS  = qS + BATCH * UDIM;                           // [32][128]
  float* vnS  = scS + BATCH * TSEQ;                          // [512]
  int* lens_s = (int*)(vnS + UDIM);                          // [32]

  const int tid  = threadIdx.x;
  const int lane = tid & 31;
  const int w    = tid >> 5;
  const int u0   = w * 16;
  const int nl   = lane & 15;
  const int mh   = lane >> 4;

  for (int i = tid; i < BATCH * UDIM; i += 1024) {
    attnB[i] = 0;
    h0B[i] = f2bf(h0i[i]);
    h1B[i] = f2bf(h1i[i]);
  }
  for (int i = tid; i < UDIM; i += 1024) vnS[i] = vn[i];
  if (tid < BATCH) lens_s[tid] = lens[tid];

  float c0_r[2][8], c1_r[2][8];
#pragma unroll
  for (int mt = 0; mt < 2; mt++)
#pragma unroll
    for (int j = 0; j < 8; j++) {
      int b = mt * 16 + mh * 8 + j;
      c0_r[mt][j] = c0i[b * UDIM + u0 + nl];
      c1_r[mt][j] = c1i[b * UDIM + u0 + nl];
    }
  float b1v[4];
#pragma unroll
  for (int g = 0; g < 4; g++) b1v[g] = bias1[g * UDIM + u0 + nl];
  __syncthreads();

  float vv[16];
#pragma unroll
  for (int i = 0; i < 16; i++) vv[i] = vnS[lane * 16 + i];
  const int b_w = w;                      // wave's batch row for attention math
  const int Lw  = lens_s[b_w];

  for (int t = 0; t < TSEQ; t++) {
    // ---- layer0: z = zx0[t] + [attn | h0] @ W0 ----
    v8f acc[4][2];
#pragma unroll
    for (int g = 0; g < 4; g++) { acc[g][0] = zero8(); acc[g][1] = zero8(); }
    for (int kk = 0; kk < 1024; kk += 32) {
      v16bf a0 = load_a_frag_split(attnB, UDIM, h0B, UDIM, UDIM, 0,  kk, lane);
      v16bf a1 = load_a_frag_split(attnB, UDIM, h0B, UDIM, UDIM, 16, kk, lane);
#pragma unroll
      for (int g = 0; g < 4; g++) {
        v16bf bf = load_b_frag(W0r, 1024, g * UDIM + u0, kk, lane);
        acc[g][0] = wmma_bf16(a0, bf, acc[g][0]);
        acc[g][1] = wmma_bf16(a1, bf, acc[g][1]);
      }
    }
    float h0n[2][8];
#pragma unroll
    for (int mt = 0; mt < 2; mt++)
#pragma unroll
      for (int j = 0; j < 8; j++) {
        int b = mt * 16 + mh * 8 + j;
        int zidx = ((b << 7) + t) * 2048 + u0 + nl;
        float gi = acc[0][mt][j] + zx0[zidx];
        float gj = acc[1][mt][j] + zx0[zidx + 512];
        float gf = acc[2][mt][j] + zx0[zidx + 1024];
        float go = acc[3][mt][j] + zx0[zidx + 1536];
        float cn = sigm(gf + 1.0f) * c0_r[mt][j] + sigm(gi) * tanhf(gj);
        c0_r[mt][j] = cn;
        h0n[mt][j] = sigm(go) * tanhf(cn);
      }
    __syncthreads();
#pragma unroll
    for (int mt = 0; mt < 2; mt++)
#pragma unroll
      for (int j = 0; j < 8; j++) {
        int b = mt * 16 + mh * 8 + j;
        h0B[b * UDIM + u0 + nl] = f2bf(h0n[mt][j]);
      }
    __syncthreads();

    // ---- layer1: z = [h0n | h1] @ W1 + bias1 ----
#pragma unroll
    for (int g = 0; g < 4; g++) { acc[g][0] = zero8(); acc[g][1] = zero8(); }
    for (int kk = 0; kk < 1024; kk += 32) {
      v16bf a0 = load_a_frag_split(h0B, UDIM, h1B, UDIM, UDIM, 0,  kk, lane);
      v16bf a1 = load_a_frag_split(h0B, UDIM, h1B, UDIM, UDIM, 16, kk, lane);
#pragma unroll
      for (int g = 0; g < 4; g++) {
        v16bf bf = load_b_frag(W1r, 1024, g * UDIM + u0, kk, lane);
        acc[g][0] = wmma_bf16(a0, bf, acc[g][0]);
        acc[g][1] = wmma_bf16(a1, bf, acc[g][1]);
      }
    }
    float h1n[2][8];
#pragma unroll
    for (int mt = 0; mt < 2; mt++)
#pragma unroll
      for (int j = 0; j < 8; j++) {
        float gi = acc[0][mt][j] + b1v[0];
        float gj = acc[1][mt][j] + b1v[1];
        float gf = acc[2][mt][j] + b1v[2];
        float go = acc[3][mt][j] + b1v[3];
        float cn = sigm(gf + 1.0f) * c1_r[mt][j] + sigm(gi) * tanhf(gj);
        c1_r[mt][j] = cn;
        h1n[mt][j] = sigm(go) * tanhf(cn);
      }
    __syncthreads();
#pragma unroll
    for (int mt = 0; mt < 2; mt++)
#pragma unroll
      for (int j = 0; j < 8; j++) {
        int b = mt * 16 + mh * 8 + j;
        h1B[b * UDIM + u0 + nl] = f2bf(h1n[mt][j]);
      }
    __syncthreads();

    // ---- query: q = h1n @ w_query ----
    v8f qa[2];
    qa[0] = zero8(); qa[1] = zero8();
    for (int kk = 0; kk < UDIM; kk += 32) {
      v16bf a0 = load_a_frag(h1B, UDIM, 0,  kk, lane);
      v16bf a1 = load_a_frag(h1B, UDIM, 16, kk, lane);
      v16bf bq = load_b_frag(Wqr, UDIM, u0, kk, lane);
      qa[0] = wmma_bf16(a0, bq, qa[0]);
      qa[1] = wmma_bf16(a1, bq, qa[1]);
    }
#pragma unroll
    for (int mt = 0; mt < 2; mt++)
#pragma unroll
      for (int j = 0; j < 8; j++)
        qS[(mt * 16 + mh * 8 + j) * UDIM + u0 + nl] = qa[mt][j];
    __syncthreads();

    // ---- score: v . tanh(keys + q)  (wave b_w, lanes split u) ----
    float qv[16];
#pragma unroll
    for (int i = 0; i < 16; i++) qv[i] = qS[b_w * UDIM + lane * 16 + i];
    for (int tin = 0; tin < TSEQ; tin++) {
      const float* kp = keys + (size_t)((b_w << 7) + tin) * UDIM + lane * 16;
      float s = 0.f;
#pragma unroll
      for (int i = 0; i < 16; i++) s += tanhf(kp[i] + qv[i]) * vv[i];
#pragma unroll
      for (int off = 16; off > 0; off >>= 1) s += __shfl_xor(s, off, 32);
      if (lane == 0) scS[b_w * TSEQ + tin] = s;
    }
    __syncthreads();

    // ---- masked softmax over tin (wave b_w) ----
    {
      float sv[4];
#pragma unroll
      for (int i = 0; i < 4; i++) {
        int tin = i * 32 + lane;
        float x = scS[b_w * TSEQ + tin];
        sv[i] = (tin < Lw) ? x : -1e9f;
      }
      float mx = fmaxf(fmaxf(sv[0], sv[1]), fmaxf(sv[2], sv[3]));
#pragma unroll
      for (int off = 16; off > 0; off >>= 1) mx = fmaxf(mx, __shfl_xor(mx, off, 32));
      float ev[4], sum = 0.f;
#pragma unroll
      for (int i = 0; i < 4; i++) {
        int tin = i * 32 + lane;
        ev[i] = (tin < Lw) ? expf(sv[i] - mx) : 0.f;
        sum += ev[i];
      }
#pragma unroll
      for (int off = 16; off > 0; off >>= 1) sum += __shfl_xor(sum, off, 32);
      float inv = 1.0f / sum;
#pragma unroll
      for (int i = 0; i < 4; i++) scS[b_w * TSEQ + i * 32 + lane] = ev[i] * inv;
    }
    __syncthreads();

    // ---- context = alpha @ enc_out  (wave b_w, lanes split d) ----
    {
      float cacc[32];
#pragma unroll
      for (int i = 0; i < 32; i++) cacc[i] = 0.f;
      for (int tin = 0; tin < TSEQ; tin++) {
        float a = scS[b_w * TSEQ + tin];
        const float* ep = enc_out + (size_t)((b_w << 7) + tin) * 1024 + lane;
#pragma unroll
        for (int i = 0; i < 32; i++) cacc[i] += a * ep[i * 32];
      }
#pragma unroll
      for (int i = 0; i < 32; i++) ctxB[b_w * 1024 + i * 32 + lane] = f2bf(cacc[i]);
    }
    __syncthreads();

    // ---- attn_new = [h1n | ctx] @ w_attn_layer ----
    v8f aa[2];
    aa[0] = zero8(); aa[1] = zero8();
    for (int kk = 0; kk < 1536; kk += 32) {
      v16bf a0 = load_a_frag_split(h1B, UDIM, ctxB, 1024, UDIM, 0,  kk, lane);
      v16bf a1 = load_a_frag_split(h1B, UDIM, ctxB, 1024, UDIM, 16, kk, lane);
      v16bf bb = load_b_frag(Walr, 1536, u0, kk, lane);
      aa[0] = wmma_bf16(a0, bb, aa[0]);
      aa[1] = wmma_bf16(a1, bb, aa[1]);
    }
#pragma unroll
    for (int mt = 0; mt < 2; mt++)
#pragma unroll
      for (int j = 0; j < 8; j++) {
        int b = mt * 16 + mh * 8 + j;
        float val = aa[mt][j];
        dec_out[(size_t)((b << 7) + t) * UDIM + u0 + nl] = val;
        attnB[b * UDIM + u0 + nl] = f2bf(val);
      }
    __syncthreads();
  }
}

// ---------------------------------------------------------------------------
// Host orchestration
// ---------------------------------------------------------------------------
extern "C" void kernel_launch(void* const* d_in, const int* in_sizes, int n_in,
                              void* d_out, int out_size, void* d_ws, size_t ws_size,
                              hipStream_t stream) {
  (void)in_sizes; (void)n_in; (void)out_size; (void)ws_size;

  const float* embed_in  = (const float*)d_in[0];
  const float* dec_embed = (const float*)d_in[1];
  const int*   lens      = (const int*)d_in[2];
  const float* enc_fw_k  = (const float*)d_in[3];
  const float* enc_fw_b  = (const float*)d_in[4];
  const float* enc_bw_k  = (const float*)d_in[5];
  const float* enc_bw_b  = (const float*)d_in[6];
  const float* dec_k0    = (const float*)d_in[7];
  const float* dec_b0    = (const float*)d_in[8];
  const float* dec_k1    = (const float*)d_in[9];
  const float* dec_b1    = (const float*)d_in[10];
  const float* w_mem     = (const float*)d_in[11];
  const float* w_query   = (const float*)d_in[12];
  const float* b_attn    = (const float*)d_in[13];
  const float* v_attn    = (const float*)d_in[14];
  const float* g_attn    = (const float*)d_in[15];
  const float* w_attn_l  = (const float*)d_in[16];
  const float* w_proj    = (const float*)d_in[17];
  const float* b_proj    = (const float*)d_in[18];
  float* logits = (float*)d_out;

  char* ws = (char*)d_ws;
  size_t off = 0;
  auto take = [&](size_t bytes) -> void* {
    void* p = (void*)(ws + off);
    off += (bytes + 255) & ~(size_t)255;
    return p;
  };

  // bf16 weights, all K-major [N][K] (TDM-friendly + contiguous B fragments)
  unsigned short* Wx_fw  = (unsigned short*)take((size_t)2048 * 512 * 2);
  unsigned short* Wx_bw  = (unsigned short*)take((size_t)2048 * 512 * 2);
  unsigned short* Wx0    = (unsigned short*)take((size_t)2048 * 512 * 2);
  unsigned short* Wmem   = (unsigned short*)take((size_t)512 * 1024 * 2);
  unsigned short* Wproj  = (unsigned short*)take((size_t)9000 * 512 * 2);
  unsigned short* Whr_fw = (unsigned short*)take((size_t)2048 * 512 * 2);
  unsigned short* Whr_bw = (unsigned short*)take((size_t)2048 * 512 * 2);
  unsigned short* W0r    = (unsigned short*)take((size_t)2048 * 1024 * 2);
  unsigned short* W1r    = (unsigned short*)take((size_t)2048 * 1024 * 2);
  unsigned short* Wqr    = (unsigned short*)take((size_t)512 * 512 * 2);
  unsigned short* Walr   = (unsigned short*)take((size_t)512 * 1536 * 2);
  float* vn      = (float*)take((size_t)512 * 4);
  float* zx_fw   = (float*)take((size_t)BATCH * TSEQ * 2048 * 4);
  float* zx_bw   = (float*)take((size_t)BATCH * TSEQ * 2048 * 4);
  float* zx0     = (float*)take((size_t)BATCH * TSEQ * 2048 * 4);
  float* enc_out = (float*)take((size_t)BATCH * TSEQ * 1024 * 4);
  float* keys    = (float*)take((size_t)BATCH * TSEQ * 512 * 4);
  float* dec_out = (float*)take((size_t)BATCH * TSEQ * 512 * 4);
  float* c_fw    = (float*)take((size_t)BATCH * UDIM * 4);
  float* h_fw    = (float*)take((size_t)BATCH * UDIM * 4);
  float* c_bw    = (float*)take((size_t)BATCH * UDIM * 4);
  float* h_bw    = (float*)take((size_t)BATCH * UDIM * 4);

  // ---- weight conversion (fp32 -> bf16, K-major) ----
  const int CB = 512;
  f32_to_bf16_T_kernel<<<CB, 256, 0, stream>>>(enc_fw_k, 2048, 0,   512,  2048, Wx_fw);
  f32_to_bf16_T_kernel<<<CB, 256, 0, stream>>>(enc_bw_k, 2048, 0,   512,  2048, Wx_bw);
  f32_to_bf16_T_kernel<<<CB, 256, 0, stream>>>(dec_k0,   2048, 0,   512,  2048, Wx0);
  f32_to_bf16_T_kernel<<<CB, 256, 0, stream>>>(w_mem,    512,  0,   1024, 512,  Wmem);
  f32_to_bf16_T_kernel<<<CB, 256, 0, stream>>>(w_proj,   9000, 0,   512,  9000, Wproj);
  f32_to_bf16_T_kernel<<<CB, 256, 0, stream>>>(enc_fw_k, 2048, 512, 512,  2048, Whr_fw);
  f32_to_bf16_T_kernel<<<CB, 256, 0, stream>>>(enc_bw_k, 2048, 512, 512,  2048, Whr_bw);
  f32_to_bf16_T_kernel<<<CB, 256, 0, stream>>>(dec_k0,   2048, 512, 1024, 2048, W0r);
  f32_to_bf16_T_kernel<<<CB, 256, 0, stream>>>(dec_k1,   2048, 0,   1024, 2048, W1r);
  f32_to_bf16_T_kernel<<<CB, 256, 0, stream>>>(w_query,  512,  0,   512,  512,  Wqr);
  f32_to_bf16_T_kernel<<<CB, 256, 0, stream>>>(w_attn_l, 512,  0,   1536, 512,  Walr);
  vnorm_kernel<<<1, 512, 0, stream>>>(v_attn, g_attn, vn);

  // ---- hoisted input-side GEMMs (x @ Wx + bias for every timestep) ----
  gemm_bf16_kernel<<<dim3(16, 64), 256, 0, stream>>>(embed_in, 512, 2048, Wx_fw, enc_fw_b, zx_fw, nullptr);
  gemm_bf16_kernel<<<dim3(16, 64), 256, 0, stream>>>(embed_in, 512, 2048, Wx_bw, enc_bw_b, zx_bw, lens);
  gemm_bf16_kernel<<<dim3(16, 64), 256, 0, stream>>>(dec_embed, 512, 2048, Wx0, dec_b0, zx0, nullptr);

  // ---- bidirectional encoder recurrence (fw + bw in parallel) ----
  encoder_kernel<<<2, 1024, 0, stream>>>(zx_fw, zx_bw, Whr_fw, Whr_bw, lens,
                                         enc_out, c_fw, h_fw, c_bw, h_bw);

  // ---- attention keys (b_attn folded as bias) ----
  gemm_bf16_kernel<<<dim3(4, 64), 256, 0, stream>>>(enc_out, 1024, 512, Wmem, b_attn, keys, nullptr);

  // ---- decoder recurrence with attention ----
  size_t smem = (size_t)(BATCH * UDIM * 2) * 3   // attn, h0, h1 (bf16)
              + (size_t)BATCH * 1024 * 2         // ctx (bf16)
              + (size_t)BATCH * UDIM * 4         // q (f32)
              + (size_t)BATCH * TSEQ * 4         // score/alpha (f32)
              + (size_t)UDIM * 4                 // v_norm
              + (size_t)BATCH * 4;               // lens
  decoder_kernel<<<1, 1024, smem, stream>>>(zx0, W0r, W1r, dec_b1, Wqr, Walr,
                                            keys, enc_out, vn, lens,
                                            c_fw, h_fw, c_bw, h_bw, dec_out);

  // ---- vocab projection -> logits ----
  gemm_bf16_kernel<<<dim3((VDIM + 127) / 128, 64), 256, 0, stream>>>(
      dec_out, 512, VDIM, Wproj, b_proj, logits, nullptr);
}
